// GraphResBlock_36120674960048
// MI455X (gfx1250) — matrix operand order repaired and checked by reading, not compiled, not run on hardware
//
#include <hip/hip_runtime.h>

// Problem constants (match reference)
#define Vn      256
#define On      128
#define En      1024
#define Dn      128
#define TWO_Dn  256
#define VOn     (Vn * On)
#define VEn     (Vn * En)
#define EPSn    1e-5f

typedef __attribute__((ext_vector_type(16))) _Float16 v16h;
typedef __attribute__((ext_vector_type(8)))  float    v8f;

// ---------------------------------------------------------------------------
// WMMA fragment loaders (wave32, layouts per CDNA5 ISA 7.12.2)
// ---------------------------------------------------------------------------
__device__ __forceinline__ v16h load_a_frag(const _Float16* __restrict__ row, int lane) {
  const int half = lane >> 4;
  v16h a;
#pragma unroll
  for (int r = 0; r < 8; ++r) {
    const int k = ((r < 4) ? 0 : 16) + 8 * half + 2 * (r & 3);
    a[2 * r]     = row[k];
    a[2 * r + 1] = row[k + 1];
  }
  return a;
}

__device__ __forceinline__ v16h load_b_frag(const _Float16* __restrict__ w, int ldk, int lane) {
  const int n  = lane & 15;
  const int kb = (lane >> 4) << 3;   // lanes 0-15: K=0..7,16..23, lanes 16-31: K=8..15,24..31
  v16h b;
#pragma unroll
  for (int r = 0; r < 8; ++r) {
    const int k = ((r < 4) ? 0 : 16) + kb + 2 * (r & 3);
    b[2 * r]     = w[n * ldk + k];
    b[2 * r + 1] = w[n * ldk + k + 1];
  }
  return b;
}

// ---------------------------------------------------------------------------
// Utility kernels
// ---------------------------------------------------------------------------
__global__ void zero_kernel(float* __restrict__ p, int n) {
  int i = blockIdx.x * blockDim.x + threadIdx.x;
  if (i < n) p[i] = 0.0f;
}

__global__ void convert_f32_f16_kernel(const float* __restrict__ src,
                                       _Float16* __restrict__ dst, int n) {
  int i = blockIdx.x * blockDim.x + threadIdx.x;
  if (i < n) dst[i] = (_Float16)src[i];
}

// Per-node src/dst appearance counts (shared by both units).
__global__ void count_edges_kernel(const long long* __restrict__ edges,
                                   float* __restrict__ cnt_src,
                                   float* __restrict__ cnt_dst) {
  int i = blockIdx.x * blockDim.x + threadIdx.x;
  if (i >= VEn) return;
  const int v = i / En;
  const int s = (int)edges[(size_t)i * 3 + 0];
  const int o = (int)edges[(size_t)i * 3 + 2];
  atomicAdd(&cnt_src[v * On + s], 1.0f);
  atomicAdd(&cnt_dst[v * On + o], 1.0f);
}

// BN1 stats over gathered rows computed as count-weighted node sums.
__global__ void __launch_bounds__(128)
bn1_stats_kernel(const float* __restrict__ x, const float* __restrict__ cs,
                 const float* __restrict__ cd, float* __restrict__ sum,
                 float* __restrict__ sq) {
  const int v = blockIdx.x;
  const int d = threadIdx.x;  // 0..127
  const float* xb  = x  + (size_t)v * On * Dn;
  const float* csb = cs + (size_t)v * On;
  const float* cdb = cd + (size_t)v * On;
  float as = 0.f, as2 = 0.f, ad = 0.f, ad2 = 0.f;
  for (int n = 0; n < On; ++n) {
    const float val = xb[n * Dn + d];
    const float ws = csb[n], wd = cdb[n];
    as += ws * val;  as2 += ws * val * val;
    ad += wd * val;  ad2 += wd * val * val;
  }
  atomicAdd(&sum[d], as);        atomicAdd(&sq[d], as2);
  atomicAdd(&sum[Dn + d], ad);   atomicAdd(&sq[Dn + d], ad2);
}

__global__ void bn_finalize_kernel(const float* __restrict__ sum,
                                   const float* __restrict__ sq, float inv_count,
                                   float* __restrict__ mean, float* __restrict__ istd,
                                   int n) {
  int d = blockIdx.x * blockDim.x + threadIdx.x;
  if (d >= n) return;
  const float m   = sum[d] * inv_count;
  const float var = sq[d] * inv_count - m * m;
  mean[d] = m;
  istd[d] = rsqrtf(var + EPSn);
}

// Produce src-normalized and dst-normalized ReLU'd f16 feature copies.
__global__ void normalize_x_kernel(const float* __restrict__ x,
                                   const float* __restrict__ mean,
                                   const float* __restrict__ istd,
                                   const float* __restrict__ g1,
                                   const float* __restrict__ b1,
                                   _Float16* __restrict__ xs,
                                   _Float16* __restrict__ xd) {
  int i = blockIdx.x * blockDim.x + threadIdx.x;
  if (i >= VOn * Dn) return;
  const int d = i & (Dn - 1);
  const float val = x[i];
  const float s = fmaxf((val - mean[d])       * istd[d]       * g1[d]       + b1[d],       0.f);
  const float t = fmaxf((val - mean[Dn + d])  * istd[Dn + d]  * g1[Dn + d]  + b1[Dn + d],  0.f);
  xs[i] = (_Float16)s;
  xd[i] = (_Float16)t;
}

// BN2 stats over pooled node features (unweighted).
__global__ void __launch_bounds__(128)
bn2_stats_kernel(const float* __restrict__ p, float* __restrict__ sum,
                 float* __restrict__ sq) {
  const int v = blockIdx.x;
  const int d = threadIdx.x;
  const float* pb = p + (size_t)v * On * Dn;
  float a = 0.f, a2 = 0.f;
  for (int n = 0; n < On; ++n) {
    const float val = pb[n * Dn + d];
    a += val;  a2 += val * val;
  }
  atomicAdd(&sum[d], a);
  atomicAdd(&sq[d], a2);
}

// ---------------------------------------------------------------------------
// Main edge kernel: one workgroup per batch v (one WGP; full LDS working set).
// Dynamic LDS: xs tile 32K + xd tile 32K + W1 64K + edge indices 8K = 136 KB
// Static  LDS: pooled f32 accumulator 64 KB (ds_add_f32 scatter target)
// 8 waves; each wave: 16-edge tile, A(16x256) x B(256x128) via
// v_wmma_f32_16x16x32_f16; B fragments stay register-resident across tiles
// (compiler uses gfx1250 VGPR-MSB addressing), A comes from ds_load_b128.
// ---------------------------------------------------------------------------
__global__ void __launch_bounds__(256)
edge_gemm_pool_kernel(const _Float16* __restrict__ xs,
                      const _Float16* __restrict__ xd,
                      const _Float16* __restrict__ w1h,
                      const float* __restrict__ lb1,
                      const long long* __restrict__ edges,
                      const float* __restrict__ cnt_src,
                      float* __restrict__ pooled) {
  extern __shared__ __align__(16) _Float16 dynsmem[];
  _Float16* sxs  = dynsmem;                              // On*Dn halves
  _Float16* sxd  = dynsmem + On * Dn;                    // On*Dn halves
  _Float16* sw1  = dynsmem + 2 * On * Dn;                // Dn*TWO_Dn halves
  int*      sidx = (int*)(dynsmem + 2 * On * Dn + Dn * TWO_Dn); // En src idx
  int*      oidx = sidx + En;                            // En dst idx
  __shared__ float spool[On * Dn];                       // 64 KB static

  const int v    = blockIdx.x;
  const int tid  = threadIdx.x;
  const int lane = tid & 31;
  const int wave = tid >> 5;

  // Cooperative staging.
  {
    const uint4* gs = (const uint4*)(xs + (size_t)v * On * Dn);
    const uint4* gd = (const uint4*)(xd + (size_t)v * On * Dn);
    const uint4* gw = (const uint4*)w1h;
    uint4* ls = (uint4*)sxs;
    uint4* ld = (uint4*)sxd;
    uint4* lw = (uint4*)sw1;
    for (int i = tid; i < On * Dn / 8; i += 256) { ls[i] = gs[i]; ld[i] = gd[i]; }
    for (int i = tid; i < Dn * TWO_Dn / 8; i += 256) lw[i] = gw[i];
    const long long* ev = edges + (size_t)v * En * 3;
    for (int e = tid; e < En; e += 256) {
      sidx[e] = (int)ev[(size_t)e * 3 + 0];
      oidx[e] = (int)ev[(size_t)e * 3 + 2];
    }
    for (int i = tid; i < On * Dn; i += 256) spool[i] = 0.0f;
  }
  __syncthreads();

  const int m_self = lane & 15;
  const int half   = lane >> 4;

  // 64 tiles of 16 edges; wave-strided. All loop memory traffic is LDS.
  for (int t = wave; t < En / 16; t += 8) {
    const int e0 = t * 16;
    const int sI = sidx[e0 + m_self];
    const int oI = oidx[e0 + m_self];

    v8f acc[8] = {};

#pragma unroll
    for (int kk = 0; kk < 8; ++kk) {  // K = 256 in chunks of 32
      const _Float16* arow = (kk < 4) ? (sxs + (size_t)sI * Dn + kk * 32)
                                      : (sxd + (size_t)oI * Dn + (kk - 4) * 32);
      const v16h a = load_a_frag(arow, lane);
#pragma unroll
      for (int nt = 0; nt < 8; ++nt) {
        const v16h b = load_b_frag(sw1 + (size_t)(nt * 16) * TWO_Dn + kk * 32, TWO_Dn, lane);
        acc[nt] = __builtin_amdgcn_wmma_f32_16x16x32_f16(
            false, a, false, b, (short)0, acc[nt], false, false);
      }
    }

    // Scatter-add to batch-local pooled accumulator (ds_add_f32).
    // C layout: VGPR r -> M = r + 8*half, N = nt*16 + (lane&15).
    // Bias lb1 is folded into the finalize (cnt * lb1), not added here.
#pragma unroll
    for (int r = 0; r < 8; ++r) {
      const int m    = r + 8 * half;
      const int srow = __shfl(sI, m, 32);
#pragma unroll
      for (int nt = 0; nt < 8; ++nt) {
        const int n = nt * 16 + m_self;
        atomicAdd(&spool[srow * Dn + n], acc[nt][r]);
      }
    }
  }
  __syncthreads();

  // Mean pooling: add per-edge bias contribution (cnt * lb1), divide by count.
  float* gp = pooled + (size_t)v * On * Dn;
  const float* cs = cnt_src + (size_t)v * On;
  for (int i = tid; i < On * Dn; i += 256) {
    const float c = cs[i / Dn];
    gp[i] = (spool[i] + c * lb1[i & (Dn - 1)]) / fmaxf(c, 1.0f);
  }
}

// ---------------------------------------------------------------------------
// Second GEMM per batch: out = relu(bn2(pooled)) @ W2^T + lb2 (+ residual)
// 128x128x128, 8 waves each own one 16-row M tile.
// ---------------------------------------------------------------------------
__global__ void __launch_bounds__(256)
node_gemm_out_kernel(const float* __restrict__ pooled,
                     const float* __restrict__ mean, const float* __restrict__ istd,
                     const float* __restrict__ g2, const float* __restrict__ b2,
                     const _Float16* __restrict__ w2h, const float* __restrict__ lb2,
                     const float* __restrict__ residual, float* __restrict__ out) {
  __shared__ _Float16 sa[On * Dn];  // 32 KB: normalized activations
  __shared__ _Float16 sw[Dn * Dn];  // 32 KB: W2 (row-major, [n][k])

  const int v   = blockIdx.x;
  const int tid = threadIdx.x;

  const float* gp = pooled + (size_t)v * On * Dn;
  for (int i = tid; i < On * Dn; i += 256) {
    const int d = i & (Dn - 1);
    const float val = (gp[i] - mean[d]) * istd[d] * g2[d] + b2[d];
    sa[i] = (_Float16)fmaxf(val, 0.0f);
  }
  {
    const uint4* gw = (const uint4*)w2h;
    uint4* lw = (uint4*)sw;
    for (int i = tid; i < Dn * Dn / 8; i += 256) lw[i] = gw[i];
  }
  __syncthreads();

  const int lane   = tid & 31;
  const int wave   = tid >> 5;
  const int m_self = lane & 15;
  const int half   = lane >> 4;
  const int mrow0  = wave * 16;

  float lbv[8];
#pragma unroll
  for (int nt = 0; nt < 8; ++nt) lbv[nt] = lb2[nt * 16 + m_self];

  v8f acc[8] = {};
#pragma unroll
  for (int kk = 0; kk < 4; ++kk) {  // K = 128 in chunks of 32
    const v16h a = load_a_frag(sa + (size_t)(mrow0 + m_self) * Dn + kk * 32, lane);
#pragma unroll
    for (int nt = 0; nt < 8; ++nt) {
      const v16h b = load_b_frag(sw + (size_t)(nt * 16) * Dn + kk * 32, Dn, lane);
      acc[nt] = __builtin_amdgcn_wmma_f32_16x16x32_f16(
          false, a, false, b, (short)0, acc[nt], false, false);
    }
  }

  float* go = out + (size_t)v * On * Dn;
  const float* gr = residual ? residual + (size_t)v * On * Dn : nullptr;
#pragma unroll
  for (int r = 0; r < 8; ++r) {
    const int m = mrow0 + r + 8 * half;
#pragma unroll
    for (int nt = 0; nt < 8; ++nt) {
      const int n = nt * 16 + m_self;
      float val = acc[nt][r] + lbv[nt];
      if (gr) val += gr[(size_t)m * Dn + n];
      go[(size_t)m * Dn + n] = val;
    }
  }
}

// ---------------------------------------------------------------------------
// Host-side orchestration
// ---------------------------------------------------------------------------
extern "C" void kernel_launch(void* const* d_in, const int* in_sizes, int n_in,
                              void* d_out, int out_size, void* d_ws, size_t ws_size,
                              hipStream_t stream) {
  const float*      obj   = (const float*)d_in[0];
  const long long*  edges = (const long long*)d_in[1];
  const float* U[2][8];
  for (int u = 0; u < 2; ++u)
    for (int p = 0; p < 8; ++p)
      U[u][p] = (const float*)d_in[2 + u * 8 + p];
  float* out = (float*)d_out;

  // Workspace carve-up (~51 MB total).
  char* ws = (char*)d_ws;
  float*    cnt_src  = (float*)ws;                 // VOn
  float*    cnt_dst  = cnt_src + VOn;              // VOn
  float*    bn1_sum  = cnt_dst + VOn;              // 256
  float*    bn1_sq   = bn1_sum + TWO_Dn;           // 256
  float*    bn1_mean = bn1_sq  + TWO_Dn;           // 256
  float*    bn1_istd = bn1_mean + TWO_Dn;          // 256
  float*    bn2_sum  = bn1_istd + TWO_Dn;          // 128
  float*    bn2_sq   = bn2_sum + Dn;               // 128
  float*    bn2_mean = bn2_sq  + Dn;               // 128
  float*    bn2_istd = bn2_mean + Dn;              // 128
  _Float16* w1h      = (_Float16*)(bn2_istd + Dn); // Dn*TWO_Dn
  _Float16* w2h      = w1h + Dn * TWO_Dn;          // Dn*Dn
  _Float16* xs       = w2h + Dn * Dn;              // VOn*Dn
  _Float16* xd       = xs + (size_t)VOn * Dn;      // VOn*Dn
  float*    pooled   = (float*)(xd + (size_t)VOn * Dn); // VOn*Dn
  float*    y0       = pooled + (size_t)VOn * Dn;       // VOn*Dn

  // Edge in/out counts (identical for both units).
  zero_kernel<<<(2 * VOn + 255) / 256, 256, 0, stream>>>(cnt_src, 2 * VOn);
  count_edges_kernel<<<VEn / 256, 256, 0, stream>>>(edges, cnt_src, cnt_dst);

  const size_t edge_lds = (size_t)(2 * On * Dn + Dn * TWO_Dn) * sizeof(_Float16)
                        + (size_t)2 * En * sizeof(int);  // 136 KB dynamic

  for (int u = 0; u < 2; ++u) {
    const float* x   = (u == 0) ? obj : y0;
    const float* g1  = U[u][0], *b1  = U[u][1], *W1 = U[u][2], *lb1 = U[u][3];
    const float* g2  = U[u][4], *b2  = U[u][5], *W2 = U[u][6], *lb2 = U[u][7];

    zero_kernel<<<2, 256, 0, stream>>>(bn1_sum, 2 * TWO_Dn);  // bn1 sum+sq
    zero_kernel<<<1, 256, 0, stream>>>(bn2_sum, 2 * Dn);      // bn2 sum+sq

    bn1_stats_kernel<<<Vn, 128, 0, stream>>>(x, cnt_src, cnt_dst, bn1_sum, bn1_sq);
    bn_finalize_kernel<<<1, 256, 0, stream>>>(bn1_sum, bn1_sq, 1.0f / (float)VEn,
                                              bn1_mean, bn1_istd, TWO_Dn);
    normalize_x_kernel<<<(VOn * Dn) / 256, 256, 0, stream>>>(x, bn1_mean, bn1_istd,
                                                             g1, b1, xs, xd);
    convert_f32_f16_kernel<<<(Dn * TWO_Dn) / 256, 256, 0, stream>>>(W1, w1h, Dn * TWO_Dn);
    convert_f32_f16_kernel<<<(Dn * Dn) / 256, 256, 0, stream>>>(W2, w2h, Dn * Dn);

    edge_gemm_pool_kernel<<<Vn, 256, edge_lds, stream>>>(xs, xd, w1h, lb1, edges,
                                                         cnt_src, pooled);

    bn2_stats_kernel<<<Vn, 128, 0, stream>>>(pooled, bn2_sum, bn2_sq);
    bn_finalize_kernel<<<1, 128, 0, stream>>>(bn2_sum, bn2_sq, 1.0f / (float)VOn,
                                              bn2_mean, bn2_istd, Dn);

    node_gemm_out_kernel<<<Vn, 256, 0, stream>>>(pooled, bn2_mean, bn2_istd, g2, b2,
                                                 w2h, lb2,
                                                 (u == 0) ? nullptr : obj,
                                                 (u == 0) ? y0 : out);
  }
}
